// GSL4uu_43834436223326
// MI455X (gfx1250) — matrix operand order repaired and checked by reading, not compiled, not run on hardware
//
#include <hip/hip_runtime.h>
#include <hip/hip_bf16.h>
#include <math.h>

#define N_TOTAL  50000
#define NUM_NODE 20000
#define KDEG     32
#define CNUM     2000
#define DDIM     256
#define M_HEAD   4
#define EPSF     1e-8f

typedef __attribute__((ext_vector_type(16))) _Float16 v16h;
typedef __attribute__((ext_vector_type(8)))  float    v8f;

// ---------------------------------------------------------------------------
// WMMA fragment loaders (wave32, 16x16x32 f16).
// A (16xK=32, M x K): lane m = l&15, K-half h2 = l>>4.
//   elems 0..15 = K[h2*16 .. h2*16+15]
// B (K=32 x 16, K x N): lane n = l&15; elems 0..15 = K[h2*16 .. h2*16+15].
// Both operands stored K-contiguous per row/column.
// ---------------------------------------------------------------------------
__device__ __forceinline__ v16h load_frag_a(const _Float16* __restrict__ row,
                                            int kbase, int h2) {
  v16h a;
#pragma unroll
  for (int i = 0; i < 16; ++i) a[i] = row[kbase + h2 * 16 + i];
  return a;
}

__device__ __forceinline__ v16h load_frag_b(const _Float16* __restrict__ col,
                                            int kbase, int h2) {
  v16h b;
#pragma unroll
  for (int e = 0; e < 16; ++e) b[e] = col[kbase + h2 * 16 + e];
  return b;
}

// ---------------------------------------------------------------------------
// Kernel 0: W[h][k][n] (f32) -> Wt[h][n][k] (f16), N-major so B columns are
// contiguous along K.
// ---------------------------------------------------------------------------
__global__ void gsl_prep_w(const float* __restrict__ W, _Float16* __restrict__ Wt) {
  long long i = (long long)blockIdx.x * 256 + threadIdx.x; // 4*256*256 total
  int h = (int)(i >> 16);
  int rem = (int)(i & 65535);
  int n = rem >> 8;
  int k = rem & 255;
  Wt[i] = (_Float16)W[((long long)h << 16) + (long long)k * 256 + n];
}

// ---------------------------------------------------------------------------
// Kernel 1: per (tile of 16 gathered rows, head h):
//   z = relu(x @ W[h] + b[h]);  out = f16( z / (||z|| + eps) )
// block = 128 threads (4 waves). Each wave owns 64 output columns (4 n-tiles).
// ---------------------------------------------------------------------------
__global__ void gsl_transform(const float* __restrict__ emb,
                              const int* __restrict__ idx, int count,
                              const _Float16* __restrict__ Wt,  // [H][256][256]
                              const float* __restrict__ bias,   // [H][256]
                              _Float16* __restrict__ out) {     // [H][count][256]
  __shared__ _Float16 lA[16][DDIM];
  __shared__ float    lZ[16][DDIM];
  __shared__ float    lP[16][8];
  __shared__ float    lInv[16];

  const int tile = blockIdx.x;
  const int h    = blockIdx.y;
  const int t    = threadIdx.x;           // 0..127
  const int lane = t & 31, wave = t >> 5; // 4 waves
  const int mlane = lane & 15, h2 = lane >> 4;

  // Stage 16 gathered embedding rows as f16 into LDS.
  for (int e = t; e < 16 * DDIM; e += 128) {
    int r = e >> 8, c = e & 255;
    int g = idx[tile * 16 + r];
    lA[r][c] = (_Float16)emb[(long long)g * DDIM + c];
  }
  __syncthreads();

  const _Float16* Wh = Wt + (long long)h * DDIM * DDIM;
  const _Float16* Ar = &lA[mlane][0];

#pragma unroll
  for (int nt = 0; nt < 4; ++nt) {
    const int ncol = wave * 64 + nt * 16 + mlane; // lane's B/D column
    const _Float16* Bc = Wh + (long long)ncol * DDIM;
    v8f acc = {};
#pragma unroll
    for (int kk = 0; kk < 8; ++kk) {
      v16h a = load_frag_a(Ar, kk * 32, h2);
      v16h b = load_frag_b(Bc, kk * 32, h2);
      acc = __builtin_amdgcn_wmma_f32_16x16x32_f16(
          false, a, false, b, (short)0, acc, false, false);
    }
    const float bv = bias[h * DDIM + ncol];
#pragma unroll
    for (int r = 0; r < 8; ++r) {           // D: lane holds rows r+8*h2, col ncol
      float z = acc[r] + bv;
      lZ[r + 8 * h2][ncol] = z > 0.f ? z : 0.f;
    }
  }
  __syncthreads();

  // Row L2 norms: 8 partial sums per row, then reduce.
  {
    int r = t >> 3, seg = t & 7;
    float s = 0.f;
#pragma unroll
    for (int c = 0; c < 32; ++c) {
      float z = lZ[r][seg * 32 + c];
      s += z * z;
    }
    lP[r][seg] = s;
  }
  __syncthreads();
  if (t < 16) {
    float s = 0.f;
#pragma unroll
    for (int j = 0; j < 8; ++j) s += lP[t][j];
    lInv[t] = 1.f / (sqrtf(s) + EPSF);
  }
  __syncthreads();

  _Float16* orow = out + ((long long)h * count + (long long)tile * 16) * DDIM;
  for (int e = t; e < 16 * DDIM; e += 128) {
    int r = e >> 8, c = e & 255;
    orow[(long long)r * DDIM + c] = (_Float16)(lZ[r][c] * lInv[r]);
  }
}

// ---------------------------------------------------------------------------
// Kernel 2: sim_add = (1/4) * sum_h zn_h @ yn_h^T  -> out [NUM_NODE][CNUM]
// block = 256 threads (8 waves). M-tile 32, N-tile 64; each wave one 16x16.
// ---------------------------------------------------------------------------
__global__ void gsl_simadd(const _Float16* __restrict__ zn,  // [H][NUM_NODE][256]
                           const _Float16* __restrict__ yn,  // [H][CNUM][256]
                           float* __restrict__ out) {
  __shared__ _Float16 lA[32][DDIM];  // 16 KB
  __shared__ _Float16 lB[64][DDIM];  // 32 KB

  const int t = threadIdx.x;              // 0..255
  const int lane = t & 31, wave = t >> 5; // 8 waves
  const int mBase = blockIdx.y * 32;
  const int nBase = blockIdx.x * 64;
  const int mi = wave >> 2, ni = wave & 3;
  const int mlane = lane & 15, h2 = lane >> 4;

  v8f acc = {};
  for (int h = 0; h < M_HEAD; ++h) {
    const uint4* gA = (const uint4*)(zn + ((long long)h * NUM_NODE + mBase) * DDIM);
    const uint4* gB = (const uint4*)(yn + ((long long)h * CNUM + nBase) * DDIM);
    uint4* sA = (uint4*)&lA[0][0];
    uint4* sB = (uint4*)&lB[0][0];
    for (int e = t; e < 32 * 32; e += 256) sA[e] = gA[e];      // 32 rows * 32 uint4
    for (int e = t; e < 64 * 32; e += 256) {                   // 64 rows * 32 uint4
      int row = e >> 5;
      uint4 v = make_uint4(0u, 0u, 0u, 0u);
      if (nBase + row < CNUM) v = gB[e];
      sB[e] = v;
    }
    __syncthreads();

    const _Float16* Ar = &lA[mi * 16 + mlane][0];
    const _Float16* Bc = &lB[ni * 16 + mlane][0];
#pragma unroll
    for (int kk = 0; kk < 8; ++kk) {
      v16h a = load_frag_a(Ar, kk * 32, h2);
      v16h b = load_frag_b(Bc, kk * 32, h2);
      acc = __builtin_amdgcn_wmma_f32_16x16x32_f16(
          false, a, false, b, (short)0, acc, false, false);
    }
    __syncthreads();
  }

#pragma unroll
  for (int r = 0; r < 8; ++r) {
    int row = mBase + mi * 16 + r + 8 * h2;
    int col = nBase + ni * 16 + mlane;
    if (col < CNUM)
      out[(long long)row * CNUM + col] = acc[r] * 0.25f;
  }
}

// ---------------------------------------------------------------------------
// Kernel 3: per-edge cosine: sim_dele[e] = (1/4) * sum_h zn_h[src].zn_h[dst]
// One wave per edge; 8 f16 (one uint4) per lane per head.
// ---------------------------------------------------------------------------
__global__ void gsl_edge(const _Float16* __restrict__ zn,
                         const int* __restrict__ edge_dst,
                         float* __restrict__ out_dele) {
  const int gid  = blockIdx.x * blockDim.x + threadIdx.x;
  const int e    = gid >> 5;
  const int lane = threadIdx.x & 31;
  if (e >= NUM_NODE * KDEG) return;
  const int src = e >> 5;               // uniform degree K=32
  const int dst = edge_dst[e];

  float s = 0.f;
#pragma unroll
  for (int h = 0; h < M_HEAD; ++h) {
    const uint4* ps = (const uint4*)(zn + ((long long)h * NUM_NODE + src) * DDIM);
    const uint4* pd = (const uint4*)(zn + ((long long)h * NUM_NODE + dst) * DDIM);
    uint4 a = ps[lane];                 // 256 f16 = 32 uint4 per row
    uint4 b = pd[lane];
    const _Float16* af = (const _Float16*)&a;
    const _Float16* bf = (const _Float16*)&b;
#pragma unroll
    for (int i = 0; i < 8; ++i) s += (float)af[i] * (float)bf[i];
  }
#pragma unroll
  for (int off = 16; off > 0; off >>= 1) s += __shfl_xor(s, off, 32);
  if (lane == 0) out_dele[e] = s * 0.25f;
}

// ---------------------------------------------------------------------------
// Kernel 4: top-k over 32-wide rows, exact stable ranks via shuffles, in place.
// ---------------------------------------------------------------------------
__global__ void gsl_topk_dele(const float* __restrict__ prob_dele,
                              float* __restrict__ io) {
  const int gid  = blockIdx.x * blockDim.x + threadIdx.x;
  const int row  = gid >> 5;
  const int lane = threadIdx.x & 31;
  if (row >= NUM_NODE) return;
  float v = io[(long long)row * KDEG + lane];
  float p = prob_dele[row];
  float pm = p < 0.5f ? p : 0.5f;
  int k = (int)floorf((float)KDEG * (1.f - pm));
  int rank = 0;
#pragma unroll
  for (int j = 0; j < KDEG; ++j) {
    float vj = __shfl(v, j, 32);
    rank += (vj > v) || (vj == v && j < lane);
  }
  io[(long long)row * KDEG + lane] = (rank < k) ? v : 0.f;
}

// ---------------------------------------------------------------------------
// Kernel 5: top-k over 2000-wide rows, in place. 256-bin LDS radix select of
// the k-th largest orderable key, then mask (stable tie-break on equals only).
// ---------------------------------------------------------------------------
__global__ void gsl_topk_add(const float* __restrict__ prob_add,
                             float* __restrict__ io) {
  __shared__ unsigned lKey[CNUM];
  __shared__ unsigned hist[256];
  __shared__ unsigned sPrefix, sRemain;

  const int row = blockIdx.x;
  const int t   = threadIdx.x; // 256
  float* rp = io + (long long)row * CNUM;

  float p = prob_add[row];
  float pm = p < 0.05f ? p : 0.05f;
  int k = (int)floorf((float)CNUM * pm); // 0..100 (uniform over block)

  if (k <= 0) {
    for (int i = t; i < CNUM; i += 256) rp[i] = 0.f;
    return;
  }

  for (int i = t; i < CNUM; i += 256) {
    unsigned u = __float_as_uint(rp[i]);
    lKey[i] = (u & 0x80000000u) ? ~u : (u | 0x80000000u);
  }
  if (t == 0) { sPrefix = 0u; sRemain = (unsigned)k; }
  __syncthreads();

  for (int b = 3; b >= 0; --b) {
    hist[t] = 0u;
    __syncthreads();
    const unsigned pre = sPrefix;
    const unsigned hmask = (b == 3) ? 0u : (0xFFFFFFFFu << ((b + 1) * 8));
    for (int i = t; i < CNUM; i += 256) {
      unsigned key = lKey[i];
      if ((key & hmask) == pre)
        atomicAdd(&hist[(key >> (b * 8)) & 255u], 1u);
    }
    __syncthreads();
    if (t == 0) {
      unsigned need = sRemain, cum = 0u;
      int chosen = 0;
      for (int bin = 255; bin >= 0; --bin) {
        unsigned c = hist[bin];
        if (cum + c >= need) { chosen = bin; sRemain = need - cum; break; }
        cum += c;
      }
      sPrefix = pre | ((unsigned)chosen << (b * 8));
    }
    __syncthreads();
  }

  const unsigned T = sPrefix;
  const unsigned remainEq = sRemain; // how many key==T to keep (index order)
  for (int i = t; i < CNUM; i += 256) {
    unsigned key = lKey[i];
    bool keep = key > T;
    if (key == T) {
      int eq = 0;
      for (int j = 0; j < i; ++j) eq += (lKey[j] == T); // ties are rare
      keep = ((unsigned)eq < remainEq);
    }
    unsigned u = (key & 0x80000000u) ? (key ^ 0x80000000u) : ~key;
    rp[i] = keep ? __uint_as_float(u) : 0.f;
  }
}

// ---------------------------------------------------------------------------
// Host launcher. Workspace layout (f16 elements):
//   WtN [4*256*256] | WtC [4*256*256] | zn [4*20000*256] | yn [4*2000*256]
// total ~46.1 MB.
// ---------------------------------------------------------------------------
extern "C" void kernel_launch(void* const* d_in, const int* in_sizes, int n_in,
                              void* d_out, int out_size, void* d_ws, size_t ws_size,
                              hipStream_t stream) {
  (void)in_sizes; (void)n_in; (void)out_size; (void)ws_size;

  const float* user_emb      = (const float*)d_in[0];
  const float* W_node        = (const float*)d_in[1];
  const float* b_node        = (const float*)d_in[2];
  const float* W_cluster     = (const float*)d_in[3];
  const float* b_cluster     = (const float*)d_in[4];
  const int*   all_nodes     = (const int*)d_in[5];
  const int*   cluster_nodes = (const int*)d_in[6];
  const int*   edge_dst      = (const int*)d_in[7];
  const float* prob_dele     = (const float*)d_in[8];
  const float* prob_add      = (const float*)d_in[9];

  float* out_dele = (float*)d_out;                       // [NUM_NODE*KDEG]
  float* out_add  = out_dele + (long long)NUM_NODE * KDEG; // [NUM_NODE*CNUM]

  _Float16* ws16 = (_Float16*)d_ws;
  _Float16* WtN = ws16;
  _Float16* WtC = WtN + (long long)M_HEAD * DDIM * DDIM;
  _Float16* zn  = WtC + (long long)M_HEAD * DDIM * DDIM;
  _Float16* yn  = zn  + (long long)M_HEAD * NUM_NODE * DDIM;

  // 0) weight transpose+convert (4*256*256 = 1024 blocks of 256)
  gsl_prep_w<<<1024, 256, 0, stream>>>(W_node, WtN);
  gsl_prep_w<<<1024, 256, 0, stream>>>(W_cluster, WtC);

  // 1) head transforms + L2 normalize (WMMA)
  gsl_transform<<<dim3(NUM_NODE / 16, M_HEAD), 128, 0, stream>>>(
      user_emb, all_nodes, NUM_NODE, WtN, b_node, zn);
  gsl_transform<<<dim3(CNUM / 16, M_HEAD), 128, 0, stream>>>(
      user_emb, cluster_nodes, CNUM, WtC, b_cluster, yn);

  // 2) dense node-vs-cluster GEMM (WMMA), writes sim_add/4 directly
  gsl_simadd<<<dim3((CNUM + 63) / 64, NUM_NODE / 32), 256, 0, stream>>>(
      zn, yn, out_add);

  // 3) per-edge cosine, writes sim_dele/4
  gsl_edge<<<(NUM_NODE * KDEG) / 8, 256, 0, stream>>>(zn, edge_dst, out_dele);

  // 4/5) in-place top-k masking
  gsl_topk_dele<<<NUM_NODE / 8, 256, 0, stream>>>(prob_dele, out_dele);
  gsl_topk_add<<<NUM_NODE, 256, 0, stream>>>(prob_add, out_add);
}
